// UTango_85426899518009
// MI455X (gfx1250) — matrix operand order repaired and checked by reading, not compiled, not run on hardware
//
#include <hip/hip_runtime.h>

#define Nn 50000
#define Ee 800000
#define Hh 256
#define MAXCTX 8
#define KCTX (Hh * MAXCTX)   // 2048

typedef __attribute__((ext_vector_type(16))) _Float16 v16h;
typedef __attribute__((ext_vector_type(8)))  _Float16 v8h;
typedef __attribute__((ext_vector_type(2)))  _Float16 v2h;
typedef __attribute__((ext_vector_type(8)))  float    v8f;

// ---------------------------------------------------------------------------
// Degree / normalization kernels
// ---------------------------------------------------------------------------
__global__ void deg_init_kernel(float* __restrict__ inv) {
    int i = blockIdx.x * 256 + threadIdx.x;
    if (i < Nn) inv[i] = 1.0f;                  // self-loop contributes 1
}

__global__ void deg_acc_kernel(const int* __restrict__ ei, float* __restrict__ inv) {
    int e = blockIdx.x * 256 + threadIdx.x;
    if (e < Ee) atomicAdd(&inv[ei[Ee + e]], 1.0f);   // dst row of edge_index
}

__global__ void deg_rsqrt_kernel(float* __restrict__ inv) {
    int i = blockIdx.x * 256 + threadIdx.x;
    if (i < Nn) inv[i] = rsqrtf(fmaxf(inv[i], 1.0f));
}

// ---------------------------------------------------------------------------
// One-time per call: split fp32 weight [K][256] into transposed f16 hi/lo
// planes Wt[256][K] so each wave's B fragment is 16 contiguous halves.
// ---------------------------------------------------------------------------
__global__ void wsplit_kernel(const float* __restrict__ W,
                              _Float16* __restrict__ Whi,
                              _Float16* __restrict__ Wlo, int K) {
    int idx = blockIdx.x * 256 + threadIdx.x;   // coalesced over [K][256]
    int k = idx >> 8, n = idx & 255;
    float v = W[idx];
    _Float16 h = (_Float16)v;
    Whi[(size_t)n * K + k] = h;
    Wlo[(size_t)n * K + k] = (_Float16)(v - (float)h);
}

// ---------------------------------------------------------------------------
// Fragment / staging helpers.
// A frag (16x32 f16, contiguous k order, same as B): lane holds row (lane&15); halves e=0..15 at
// k = akb+e, akb = (lane>=16)*16.  With LDS half
// planes [16][32] this is two aligned b128 LDS loads + a concat shuffle.
// ---------------------------------------------------------------------------
__device__ __forceinline__ v16h load_afrag(const _Float16* p) {
    v8h lo = *(const v8h*)(p);        // k = akb + 0..7
    v8h hi = *(const v8h*)(p + 8);    // k = akb + 8..15
    return __builtin_shufflevector(lo, hi, 0,1,2,3,4,5,6,7,8,9,10,11,12,13,14,15);
}

// packed split of 2 consecutive f32 -> hi/lo half pairs (v_cvt_pk_f16_f32)
__device__ __forceinline__ void stage2(_Float16* ph, _Float16* pl, float2 a2) {
    _Float16 h0 = (_Float16)a2.x, h1 = (_Float16)a2.y;
    v2h hp = {h0, h1};
    v2h lp = {(_Float16)(a2.x - (float)h0), (_Float16)(a2.y - (float)h1)};
    *(v2h*)ph = hp;                   // ds_store_b32
    *(v2h*)pl = lp;                   // ds_store_b32
}

// ---------------------------------------------------------------------------
// WMMA GEMM: out[16 rows x 256 cols per block] = A[N,K] @ W[K,256]
// fp32 accuracy via split-f16: acc += Ahi*Bhi + Ahi*Blo + Alo*Bhi.
// block = 512 threads (16 waves); wave w owns cols [16w,16w+16).
// Double-buffered LDS staging: ONE barrier per K-chunk; waves 0-7 stage the
// next chunk (wave-uniform branch) while all 16 waves run WMMAs on current.
// ---------------------------------------------------------------------------
__global__ __launch_bounds__(512) void gemm_split_kernel(const float* __restrict__ A,
                                                         const _Float16* __restrict__ Bthi,
                                                         const _Float16* __restrict__ Btlo,
                                                         float* __restrict__ out, int K) {
    __shared__ __align__(16) _Float16 lah[2][16 * 32];
    __shared__ __align__(16) _Float16 lal[2][16 * 32];
    const int t    = threadIdx.x;
    const int lane = t & 31;
    const int wave = t >> 5;
    const int row0 = blockIdx.x * 16;
    const int colg = wave * 16 + (lane & 15);
    const int aoff = (lane & 15) * 32 + ((lane >> 4) * 16);  // arow*32 + akb
    const int bkb  = (lane >> 4) * 16;

    const _Float16* bhip = Bthi + (size_t)colg * K + bkb;    // 32B aligned
    const _Float16* blop = Btlo + (size_t)colg * K + bkb;

    // staging role: threads 0..255 (waves 0-7), 2 consecutive k per thread
    const bool loader = (t < 256);
    const int  lr   = (t >> 4) & 15;                         // row in tile
    const int  lk2  = (t & 15) * 2;                          // k pair base
    const int  soff = lr * 32 + lk2;
    const float* arow = A + (size_t)(row0 + lr) * K + lk2;

    const int nch = K >> 5;
    if (loader)
        stage2(&lah[0][soff], &lal[0][soff], *(const float2*)&arow[0]);
    __syncthreads();

    v8f acc = {};
    for (int i = 0; i < nch; ++i) {
        const int cur = i & 1;
        if (loader && (i + 1 < nch)) {                       // wave-uniform
            stage2(&lah[cur ^ 1][soff], &lal[cur ^ 1][soff],
                   *(const float2*)&arow[(i + 1) << 5]);
            __builtin_prefetch(&arow[(i + 2) << 5], 0, 1);
        }
        v16h ahi = load_afrag(&lah[cur][aoff]);
        v16h alo = load_afrag(&lal[cur][aoff]);
        v16h bhi = *(const v16h*)(bhip + (i << 5));          // 2x global_load_b128
        v16h blo = *(const v16h*)(blop + (i << 5));

        acc = __builtin_amdgcn_wmma_f32_16x16x32_f16(false, ahi, false, bhi, (short)0, acc, false, false);
        acc = __builtin_amdgcn_wmma_f32_16x16x32_f16(false, ahi, false, blo, (short)0, acc, false, false);
        acc = __builtin_amdgcn_wmma_f32_16x16x32_f16(false, alo, false, bhi, (short)0, acc, false, false);
        __syncthreads();                                     // single barrier/chunk
    }

    const int rbase = row0 + ((lane >> 4) * 8);              // C layout
    #pragma unroll
    for (int v = 0; v < 8; ++v)
        out[(size_t)(rbase + v) * Hh + colg] = acc[v];
}

// ---------------------------------------------------------------------------
// GCN scatter: init with self-loop term + bias, per-edge atomics, ReLU.
// ---------------------------------------------------------------------------
__global__ void scat_init_kernel(const float* __restrict__ h, const float* __restrict__ inv,
                                 const float* __restrict__ b, float* __restrict__ out) {
    int i = blockIdx.x, c = threadIdx.x * 4;
    float iv2 = inv[i] * inv[i];
    float4 hv = *(const float4*)&h[(size_t)i * Hh + c];
    float4 bv = *(const float4*)&b[c];
    float4 o  = { hv.x * iv2 + bv.x, hv.y * iv2 + bv.y,
                  hv.z * iv2 + bv.z, hv.w * iv2 + bv.w };
    *(float4*)&out[(size_t)i * Hh + c] = o;
}

__global__ void scat_edge_kernel(const int* __restrict__ ei, const float* __restrict__ h,
                                 const float* __restrict__ inv, float* __restrict__ out) {
    int e = blockIdx.x, c = threadIdx.x * 4;
    int s = ei[e], d = ei[Ee + e];
    float coef = inv[s] * inv[d];
    float4 hv = *(const float4*)&h[(size_t)s * Hh + c];
    float* o = &out[(size_t)d * Hh + c];
    atomicAdd(o + 0, hv.x * coef);
    atomicAdd(o + 1, hv.y * coef);
    atomicAdd(o + 2, hv.z * coef);
    atomicAdd(o + 3, hv.w * coef);
}

__global__ void relu_kernel(float* __restrict__ x, int n) {
    int i = blockIdx.x * 256 + threadIdx.x;
    if (i < n) x[i] = fmaxf(x[i], 0.0f);
}

// ---------------------------------------------------------------------------
// Fused: ctx = (masked gather of feat via node_label, [N,2048]) @ resize_W + b
//        out = feat * ctx.  Same double-buffered WMMA tiling; A gathered on
// the fly.  Each K=32 chunk lies inside one context slot (32 | 256).
// ---------------------------------------------------------------------------
__device__ __forceinline__ float2 gather2(const float* __restrict__ feat,
                                          const int* __restrict__ node_label,
                                          int rowg, int clen, int chunk, int lk2) {
    const int j  = chunk >> 3;                 // context slot (chunk*32 >> 8)
    const int kc = (chunk & 7) << 5;           // channel base inside slot
    const int lbl = node_label[rowg * MAXCTX + j];
    float2 z = {0.0f, 0.0f};
    if (j < clen) z = *(const float2*)&feat[(size_t)lbl * Hh + kc + lk2];
    return z;
}

__global__ __launch_bounds__(512) void resize_fused_kernel(const float* __restrict__ feat,
                                                           const int* __restrict__ node_label,
                                                           const int* __restrict__ context_len,
                                                           const _Float16* __restrict__ Bthi,
                                                           const _Float16* __restrict__ Btlo,
                                                           const float* __restrict__ bias,
                                                           float* __restrict__ out) {
    __shared__ __align__(16) _Float16 lah[2][16 * 32];
    __shared__ __align__(16) _Float16 lal[2][16 * 32];
    const int t    = threadIdx.x;
    const int lane = t & 31;
    const int wave = t >> 5;
    const int row0 = blockIdx.x * 16;
    const int colg = wave * 16 + (lane & 15);
    const int aoff = (lane & 15) * 32 + ((lane >> 4) * 16);
    const int bkb  = (lane >> 4) * 16;

    const _Float16* bhip = Bthi + (size_t)colg * KCTX + bkb;
    const _Float16* blop = Btlo + (size_t)colg * KCTX + bkb;

    const bool loader = (t < 256);
    const int  lr   = (t >> 4) & 15;
    const int  lk2  = (t & 15) * 2;
    const int  soff = lr * 32 + lk2;
    const int  rowg = row0 + lr;
    const int  clen = loader ? context_len[rowg] : 0;

    const int nch = KCTX >> 5;                 // 64 chunks
    if (loader)
        stage2(&lah[0][soff], &lal[0][soff],
               gather2(feat, node_label, rowg, clen, 0, lk2));
    __syncthreads();

    v8f acc = {};
    for (int i = 0; i < nch; ++i) {
        const int cur = i & 1;
        if (loader && (i + 1 < nch)) {
            stage2(&lah[cur ^ 1][soff], &lal[cur ^ 1][soff],
                   gather2(feat, node_label, rowg, clen, i + 1, lk2));
        }
        v16h ahi = load_afrag(&lah[cur][aoff]);
        v16h alo = load_afrag(&lal[cur][aoff]);
        v16h bhi = *(const v16h*)(bhip + (i << 5));
        v16h blo = *(const v16h*)(blop + (i << 5));

        acc = __builtin_amdgcn_wmma_f32_16x16x32_f16(false, ahi, false, bhi, (short)0, acc, false, false);
        acc = __builtin_amdgcn_wmma_f32_16x16x32_f16(false, ahi, false, blo, (short)0, acc, false, false);
        acc = __builtin_amdgcn_wmma_f32_16x16x32_f16(false, alo, false, bhi, (short)0, acc, false, false);
        __syncthreads();
    }

    const float bcol = bias[colg];
    const int rbase = row0 + ((lane >> 4) * 8);
    #pragma unroll
    for (int v = 0; v < 8; ++v) {
        const size_t idx = (size_t)(rbase + v) * Hh + colg;
        out[idx] = feat[idx] * (acc[v] + bcol);
    }
}

// ---------------------------------------------------------------------------
extern "C" void kernel_launch(void* const* d_in, const int* in_sizes, int n_in,
                              void* d_out, int out_size, void* d_ws, size_t ws_size,
                              hipStream_t stream) {
    const float* x           = (const float*)d_in[0];
    const int*   edge_index  = (const int*)d_in[1];
    const int*   node_label  = (const int*)d_in[2];
    const int*   context_len = (const int*)d_in[3];
    const float* gcn_W       = (const float*)d_in[4];
    const float* gcn_b       = (const float*)d_in[5];
    const float* resize_W    = (const float*)d_in[6];
    const float* resize_b    = (const float*)d_in[7];
    float* out = (float*)d_out;

    // workspace layout (256B aligned chunks)
    char* ws = (char*)d_ws;
    size_t off = 0;
    auto alloc = [&](size_t bytes) -> char* {
        char* p = ws + off;
        off = (off + bytes + 255) & ~(size_t)255;
        return p;
    };
    float*    inv   = (float*)   alloc((size_t)Nn * 4);
    float*    bufA  = (float*)   alloc((size_t)Nn * Hh * 4);
    float*    bufB  = (float*)   alloc((size_t)Nn * Hh * 4);
    _Float16* wgHi  = (_Float16*)alloc((size_t)Hh * Hh * 2);
    _Float16* wgLo  = (_Float16*)alloc((size_t)Hh * Hh * 2);
    _Float16* wrHi  = (_Float16*)alloc((size_t)KCTX * Hh * 2);
    _Float16* wrLo  = (_Float16*)alloc((size_t)KCTX * Hh * 2);

    dim3 b256(256);

    // weight split/transpose (cheap, L2-resident)
    wsplit_kernel<<<(Hh * Hh) / 256, b256, 0, stream>>>(gcn_W, wgHi, wgLo, Hh);
    wsplit_kernel<<<(KCTX * Hh) / 256, b256, 0, stream>>>(resize_W, wrHi, wrLo, KCTX);

    // symmetric normalization coefficients
    deg_init_kernel <<<(Nn + 255) / 256, b256, 0, stream>>>(inv);
    deg_acc_kernel  <<<(Ee + 255) / 256, b256, 0, stream>>>(edge_index, inv);
    deg_rsqrt_kernel<<<(Nn + 255) / 256, b256, 0, stream>>>(inv);

    // layer 1: h1 = x @ W ; feat1 = relu(scatter(h1) + b)
    gemm_split_kernel<<<Nn / 16, 512, 0, stream>>>(x, wgHi, wgLo, bufA, Hh);
    scat_init_kernel<<<Nn, 64, 0, stream>>>(bufA, inv, gcn_b, bufB);
    scat_edge_kernel<<<Ee, 64, 0, stream>>>(edge_index, bufA, inv, bufB);
    relu_kernel<<<((size_t)Nn * Hh + 255) / 256, b256, 0, stream>>>(bufB, Nn * Hh);

    // layer 2: h2 = feat1 @ W ; feat2 = scatter(h2) + b
    gemm_split_kernel<<<Nn / 16, 512, 0, stream>>>(bufB, wgHi, wgLo, bufA, Hh);
    scat_init_kernel<<<Nn, 64, 0, stream>>>(bufA, inv, gcn_b, bufB);
    scat_edge_kernel<<<Ee, 64, 0, stream>>>(edge_index, bufA, inv, bufB);

    // fused context gather + resize GEMM + elementwise product
    resize_fused_kernel<<<Nn / 16, 512, 0, stream>>>(bufB, node_label, context_len,
                                                     wrHi, wrLo, resize_b, out);
}
